// OFF_Unit_40338332844216
// MI455X (gfx1250) — matrix operand order repaired and checked by reading, not compiled, not run on hardware
//
#include <hip/hip_runtime.h>

// ---------------------------------------------------------------------------
// Fused OFF-unit: out = w2 · [sobelx(w1s·x+b1s) | sobely(...) | Δt(w1t·x) | x] + b2
// Single pass over x, bf16 hi/lo split WMMA (3-term) for ~fp32 accuracy.
// ---------------------------------------------------------------------------

typedef __attribute__((ext_vector_type(16))) __bf16 v16bf;
typedef __attribute__((ext_vector_type(8)))  float  v8f;

#define NTHREADS 256

#define C_IN   64
#define MID    32
#define T_DIM  16
#define H_DIM  112
#define W_DIM  112
#define OUT_C  64
#define FEAT   160

#define TILE_H 2
#define TILE_W 56
#define HALO_R (TILE_H + 2)          // 4
#define HALO_C (TILE_W + 2)          // 58
#define NQ     (HALO_R * HALO_C)     // 232 halo'd pixels
#define NQP    240                   // padded to 15 M-tiles of 16
#define NP     (TILE_H * TILE_W)     // 112 center pixels = 7 M-tiles

// LDS strides padded for bank-conflict-free fragment loads
#define XSTR   66                    // x tile: [q][ch], bf16
#define SGSTR  162                   // feat staging: [p][ch 0..159], bf16
#define SSTR   241                   // s buffer: [ch][q], f32
#define TBSTR  113                   // tb buffer: [buf][ch][p], f32
#define OSTR   113                   // out staging: [ch][p], f32
#define W1STR  66                    // w1 staging: [n][k], bf16
#define W2STR  162                   // w2 staging: [n][k], bf16

__device__ __forceinline__ unsigned short f2bf(float f) {
  unsigned int u = __float_as_uint(f);
  u += 0x7FFFu + ((u >> 16) & 1u);   // round-to-nearest-even
  return (unsigned short)(u >> 16);
}
__device__ __forceinline__ float bf2f(unsigned short h) {
  return __uint_as_float(((unsigned int)h) << 16);
}

// A-fragment (16x32 bf16): lane<16 -> M=lane, K = base+{0..7,16..23};
// lane>=16 -> M=lane-16, K = base+{8..15,24..31}. Caller folds the +8 into base.
// VGPR j holds packed pair (K=2j', 2j'+1) => dwords {0,1,2,3, 8,9,10,11}.
__device__ __forceinline__ v16bf load_afrag(const unsigned short* base) {
  union { v16bf v; unsigned int d[8]; } f;
  const unsigned int* p32 = (const unsigned int*)base;
#pragma unroll
  for (int j = 0; j < 4; ++j) { f.d[j] = p32[j]; f.d[j + 4] = p32[j + 8]; }
  return f.v;
}
// B-fragment (32x16 bf16): lane<16 -> N=lane, K=base+{0..7,16..23}; lane>=16 -> N=lane-16,
// K=base+{8..15,24..31} (caller folds +8 into base). Packed pairs -> dwords {0,1,2,3, 8,9,10,11}.
__device__ __forceinline__ v16bf load_bfrag(const unsigned short* base) {
  union { v16bf v; unsigned int d[8]; } f;
  const unsigned int* p32 = (const unsigned int*)base;
#pragma unroll
  for (int j = 0; j < 4; ++j) { f.d[j] = p32[j]; f.d[j + 4] = p32[j + 8]; }
  return f.v;
}

__device__ __forceinline__ v8f wmma_bf16(v16bf a, v16bf b, v8f c) {
  return __builtin_amdgcn_wmma_f32_16x16x32_bf16(false, a, false, b,
                                                 (short)0, c, false, false);
}

__global__ __launch_bounds__(NTHREADS, 1)
void offunit_fused_kernel(const float* __restrict__ x,
                          const float* __restrict__ w1t,
                          const float* __restrict__ w1s,
                          const float* __restrict__ b1s,
                          const float* __restrict__ w2,
                          const float* __restrict__ b2,
                          float* __restrict__ out) {
  // ---- static LDS (~277 KB total, fits 320 KB WGP limit, 1 WG/WGP) ----
  __shared__ __attribute__((aligned(16))) unsigned short XH[NQP * XSTR];
  __shared__ __attribute__((aligned(16))) unsigned short XL[NQP * XSTR];
  __shared__ __attribute__((aligned(16))) unsigned short SGH[NP * SGSTR];
  __shared__ __attribute__((aligned(16))) unsigned short SGL[NP * SGSTR];
  __shared__ __attribute__((aligned(16))) float SBUF[MID * SSTR];
  __shared__ __attribute__((aligned(16))) float TBUF[2 * MID * TBSTR];
  __shared__ __attribute__((aligned(16))) float OUTS[OUT_C * OSTR];
  __shared__ __attribute__((aligned(16))) unsigned short W1TH[MID * W1STR];
  __shared__ __attribute__((aligned(16))) unsigned short W1TL[MID * W1STR];
  __shared__ __attribute__((aligned(16))) unsigned short W1SH[MID * W1STR];
  __shared__ __attribute__((aligned(16))) unsigned short W1SL[MID * W1STR];
  __shared__ __attribute__((aligned(16))) unsigned short W2H[OUT_C * W2STR];
  __shared__ __attribute__((aligned(16))) unsigned short W2L[OUT_C * W2STR];
  __shared__ float B1S[MID];
  __shared__ float B2[OUT_C];

  const int tid  = threadIdx.x;
  const int lane = tid & 31;
  const int wv   = tid >> 5;

  const int bi  = blockIdx.x;          // 0..447
  const int b   = bi / 112;
  const int rem = bi - b * 112;
  const int h0  = (rem >> 1) * TILE_H; // 56 row tiles
  const int w0  = (rem & 1) * TILE_W;  // 2 col tiles

  // ---- stage weights/biases (hi/lo bf16 split) ----
  for (int i = tid; i < MID * C_IN; i += NTHREADS) {
    int n = i >> 6, k = i & 63;
    float v = w1t[i];
    unsigned short h = f2bf(v);
    W1TH[n * W1STR + k] = h; W1TL[n * W1STR + k] = f2bf(v - bf2f(h));
    v = w1s[i]; h = f2bf(v);
    W1SH[n * W1STR + k] = h; W1SL[n * W1STR + k] = f2bf(v - bf2f(h));
  }
  for (int i = tid; i < OUT_C * FEAT; i += NTHREADS) {
    int n = i / FEAT, k = i - n * FEAT;
    float v = w2[i];
    unsigned short h = f2bf(v);
    W2H[n * W2STR + k] = h; W2L[n * W2STR + k] = f2bf(v - bf2f(h));
  }
  if (tid < MID) B1S[tid] = b1s[tid];
  else if (tid >= 64 && tid < 64 + OUT_C) B2[tid - 64] = b2[tid - 64];
  __syncthreads();

  const int kaA = (lane >= 16) ? 8 : 0;    // A-fragment K base offset
  const int kaB = (lane >= 16) ? 8 : 0;    // B-fragment K base offset
  const int nrw = lane & 15;               // N (or M) sub-index
  const int Mo  = (lane >= 16) ? 8 : 0;    // D-fragment M offset

  // Pipelined frame loop; t==T_DIM is a drain step emitting frame 15.
  for (int t = 0; t <= T_DIM; ++t) {
    const int cur = (t == T_DIM) ? 1 : (t & 1);
    const int prv = cur ^ 1;

    // ---------------- Region A: load halo'd x[t] tile, split hi/lo ----------
    if (t < T_DIM) {
      for (int i = tid; i < C_IN * NQP; i += NTHREADS) {
        int q = i % NQP, ch = i / NQP;
        float v = 0.f;
        if (q < NQ) {
          int rr = q / HALO_C, cc = q - rr * HALO_C;
          int h = h0 + rr - 1, w = w0 + cc - 1;
          if ((unsigned)h < H_DIM && (unsigned)w < W_DIM)
            v = x[(((b * C_IN + ch) * T_DIM + t) * H_DIM + h) * W_DIM + w];
        }
        unsigned short hi = f2bf(v);
        XH[q * XSTR + ch] = hi;
        XL[q * XSTR + ch] = f2bf(v - bf2f(hi));
      }
    }
    __syncthreads();

    // ------------- Region B: GEMM1 -> tb[t] (center) and s[t] (halo) -------
    if (t < T_DIM) {
      if (wv < 7) {                    // tb = w1t·x, center pixels, no bias
        const int m  = wv;
        const int pm = m * 16 + nrw;
        const int r  = pm / TILE_W, c = pm - r * TILE_W;
        const int q  = (r + 1) * HALO_C + (c + 1);
        v8f z = {};
        v8f acc0 = z, acc1 = z;
#pragma unroll
        for (int kc = 0; kc < 2; ++kc) {
          v16bf ah = load_afrag(&XH[q * XSTR + kc * 32 + kaA]);
          v16bf al = load_afrag(&XL[q * XSTR + kc * 32 + kaA]);
          v16bf b0h = load_bfrag(&W1TH[nrw * W1STR + kc * 32 + kaB]);
          v16bf b0l = load_bfrag(&W1TL[nrw * W1STR + kc * 32 + kaB]);
          v16bf b1h = load_bfrag(&W1TH[(16 + nrw) * W1STR + kc * 32 + kaB]);
          v16bf b1l = load_bfrag(&W1TL[(16 + nrw) * W1STR + kc * 32 + kaB]);
          acc0 = wmma_bf16(ah, b0h, acc0);
          acc0 = wmma_bf16(al, b0h, acc0);
          acc0 = wmma_bf16(ah, b0l, acc0);
          acc1 = wmma_bf16(ah, b1h, acc1);
          acc1 = wmma_bf16(al, b1h, acc1);
          acc1 = wmma_bf16(ah, b1l, acc1);
        }
#pragma unroll
        for (int v = 0; v < 8; ++v) {
          int p = m * 16 + v + Mo;
          TBUF[(cur * MID + nrw) * TBSTR + p]        = acc0[v];
          TBUF[(cur * MID + 16 + nrw) * TBSTR + p]   = acc1[v];
        }
      }
      for (int mi = wv; mi < 15; mi += 8) {  // s = w1s·x + b1s over halo'd tiles
        const int q0 = mi * 16 + nrw;
        v8f z = {};
        v8f acc0 = z, acc1 = z;
#pragma unroll
        for (int kc = 0; kc < 2; ++kc) {
          v16bf ah = load_afrag(&XH[q0 * XSTR + kc * 32 + kaA]);
          v16bf al = load_afrag(&XL[q0 * XSTR + kc * 32 + kaA]);
          v16bf b0h = load_bfrag(&W1SH[nrw * W1STR + kc * 32 + kaB]);
          v16bf b0l = load_bfrag(&W1SL[nrw * W1STR + kc * 32 + kaB]);
          v16bf b1h = load_bfrag(&W1SH[(16 + nrw) * W1STR + kc * 32 + kaB]);
          v16bf b1l = load_bfrag(&W1SL[(16 + nrw) * W1STR + kc * 32 + kaB]);
          acc0 = wmma_bf16(ah, b0h, acc0);
          acc0 = wmma_bf16(al, b0h, acc0);
          acc0 = wmma_bf16(ah, b0l, acc0);
          acc1 = wmma_bf16(ah, b1h, acc1);
          acc1 = wmma_bf16(al, b1h, acc1);
          acc1 = wmma_bf16(ah, b1l, acc1);
        }
#pragma unroll
        for (int v = 0; v < 8; ++v) {
          int q  = mi * 16 + v + Mo;
          int rr = q / HALO_C, cc = q - rr * HALO_C;
          int h  = h0 + rr - 1, w = w0 + cc - 1;
          // zero-padding semantics: bias only where the padded pixel is in-image
          bool inb = (q < NQ) && ((unsigned)h < H_DIM) && ((unsigned)w < W_DIM);
          SBUF[nrw * SSTR + q]        = acc0[v] + (inb ? B1S[nrw] : 0.f);
          SBUF[(16 + nrw) * SSTR + q] = acc1[v] + (inb ? B1S[16 + nrw] : 0.f);
        }
      }
    }
    __syncthreads();

    // --------- Region C: stage dt = tb[t] - tb[t-1] into feat ch 64..95 ----
    if (t >= 1) {
      for (int i = tid; i < MID * NP; i += NTHREADS) {
        int ch = i / NP, p = i - ch * NP;
        float d = TBUF[(cur * MID + ch) * TBSTR + p]
                - TBUF[(prv * MID + ch) * TBSTR + p];
        unsigned short hi = f2bf(d);
        SGH[p * SGSTR + 64 + ch] = hi;
        SGL[p * SGSTR + 64 + ch] = f2bf(d - bf2f(hi));
      }
    }
    __syncthreads();

    // --------- Region D: GEMM2 (160 -> 64) for frame t-1 into OUTS ---------
    if (t >= 1 && wv < 7) {
      const int m  = wv;
      const int p0 = m * 16 + nrw;
      v8f z = {};
      v8f acc[4] = {z, z, z, z};
#pragma unroll
      for (int kc = 0; kc < 5; ++kc) {
        v16bf ah = load_afrag(&SGH[p0 * SGSTR + kc * 32 + kaA]);
        v16bf al = load_afrag(&SGL[p0 * SGSTR + kc * 32 + kaA]);
#pragma unroll
        for (int n = 0; n < 4; ++n) {
          v16bf bh = load_bfrag(&W2H[(n * 16 + nrw) * W2STR + kc * 32 + kaB]);
          v16bf bl = load_bfrag(&W2L[(n * 16 + nrw) * W2STR + kc * 32 + kaB]);
          acc[n] = wmma_bf16(ah, bh, acc[n]);
          acc[n] = wmma_bf16(al, bh, acc[n]);
          acc[n] = wmma_bf16(ah, bl, acc[n]);
        }
      }
#pragma unroll
      for (int n = 0; n < 4; ++n) {
        int ch = n * 16 + nrw;
#pragma unroll
        for (int v = 0; v < 8; ++v)
          OUTS[ch * OSTR + m * 16 + v + Mo] = acc[n][v] + B2[ch];
      }
    }
    __syncthreads();

    // --- Region E: coalesced store of frame t-1; stage fx/fy/x for frame t -
    if (t >= 1) {
      const int f = t - 1;
      for (int i = tid; i < OUT_C * NP; i += NTHREADS) {
        int ch = i / NP, p = i - ch * NP;
        int r = p / TILE_W, c = p - r * TILE_W;
        out[(((b * OUT_C + ch) * T_DIM + f) * H_DIM + (h0 + r)) * W_DIM +
            (w0 + c)] = OUTS[ch * OSTR + p];
      }
    }
    if (t < T_DIM) {
      // Sobel (cross-correlation, zero-pad already baked into SBUF)
      for (int i = tid; i < MID * NP; i += NTHREADS) {
        int ch = i / NP, p = i - ch * NP;
        int r = p / TILE_W, c = p - r * TILE_W;
        const float* S = &SBUF[ch * SSTR];
        int q00 = r * HALO_C + c;   // (r-1,c-1) in halo coords
        float s00 = S[q00],                 s01 = S[q00 + 1],              s02 = S[q00 + 2];
        float s10 = S[q00 + HALO_C],                                       s12 = S[q00 + HALO_C + 2];
        float s20 = S[q00 + 2 * HALO_C],    s21 = S[q00 + 2 * HALO_C + 1], s22 = S[q00 + 2 * HALO_C + 2];
        float fx = (s02 + s12 + s22) - (s00 + s10 + s20);
        float fy = (s20 + s21 + s22) - (s00 + s01 + s02);
        unsigned short hx = f2bf(fx);
        SGH[p * SGSTR + ch]      = hx; SGL[p * SGSTR + ch]      = f2bf(fx - bf2f(hx));
        unsigned short hy = f2bf(fy);
        SGH[p * SGSTR + 32 + ch] = hy; SGL[p * SGSTR + 32 + ch] = f2bf(fy - bf2f(hy));
      }
      // pass-through x (already bf16 hi/lo in X tile) -> feat ch 96..159
      for (int i = tid; i < C_IN * NP; i += NTHREADS) {
        int ch = i / NP, p = i - ch * NP;
        int r = p / TILE_W, c = p - r * TILE_W;
        int q = (r + 1) * HALO_C + (c + 1);
        SGH[p * SGSTR + 96 + ch] = XH[q * XSTR + ch];
        SGL[p * SGSTR + 96 + ch] = XL[q * XSTR + ch];
      }
    }
    __syncthreads();
  }
}

extern "C" void kernel_launch(void* const* d_in, const int* in_sizes, int n_in,
                              void* d_out, int out_size, void* d_ws, size_t ws_size,
                              hipStream_t stream) {
  (void)in_sizes; (void)n_in; (void)out_size; (void)d_ws; (void)ws_size;
  const float* x   = (const float*)d_in[0];
  const float* w1t = (const float*)d_in[1];
  // d_in[2] = b1t: cancels exactly in the temporal difference -> unused
  const float* w1s = (const float*)d_in[3];
  const float* b1s = (const float*)d_in[4];
  const float* w2  = (const float*)d_in[5];
  const float* b2  = (const float*)d_in[6];
  float* out = (float*)d_out;

  dim3 grid(4 * 56 * 2);   // B x (H/TILE_H) x (W/TILE_W) = 448 workgroups
  dim3 block(NTHREADS);
  offunit_fused_kernel<<<grid, block, 0, stream>>>(x, w1t, w1s, b1s, w2, b2, out);
}
